// GNNModel_21191368639178
// MI455X (gfx1250) — compile-verified, matrix operand order checked by reading, not run on hardware
//
#include <hip/hip_runtime.h>

typedef __attribute__((ext_vector_type(16))) __bf16 v16bf;
typedef __attribute__((ext_vector_type(8)))  __bf16 v8bf;
typedef __attribute__((ext_vector_type(8)))  float  v8f;

// ---------------------------------------------------------------------------
// degree / pooling counts
// ---------------------------------------------------------------------------
__global__ void count_edges(const int* __restrict__ dst, float* __restrict__ cnt, int E) {
    int e = blockIdx.x * blockDim.x + threadIdx.x;
    if (e < E) atomicAdd(&cnt[dst[e]], 1.0f);
}

// invc padded to Mpad rows: pad rows get 0 so padded GEMM rows contribute nothing
__global__ void inv_counts(const float* __restrict__ cnt, float* __restrict__ invc,
                           int N, int Mpad) {
    int n = blockIdx.x * blockDim.x + threadIdx.x;
    if (n < Mpad) invc[n] = (n < N) ? 1.0f / fmaxf(cnt[n], 1.0f) : 0.0f;
}

__global__ void graph_counts(const int* __restrict__ batch, float* __restrict__ gcnt, int N) {
    int n = blockIdx.x * blockDim.x + threadIdx.x;
    if (n < N) atomicAdd(&gcnt[batch[n]], 1.0f);
}

// ---------------------------------------------------------------------------
// W [K,N] fp32  ->  Wt [N,K] bf16  (per-lane-contiguous B fragments)
// ---------------------------------------------------------------------------
__global__ void convert_w(const float* __restrict__ W, __bf16* __restrict__ Wt,
                          int K, int N) {
    int tid = blockIdx.x * blockDim.x + threadIdx.x;
    if (tid >= K * N) return;
    int k = tid / N;
    int n = tid - k * N;
    Wt[(size_t)n * K + k] = (__bf16)W[tid];
}

// ---------------------------------------------------------------------------
// edge scatter:  aggr[dst] += h[src] * edge_attr   (sum part of scatter-mean)
// ---------------------------------------------------------------------------
__global__ void scatter3(const float* __restrict__ x, const int* __restrict__ src,
                         const int* __restrict__ dst, const float* __restrict__ ea,
                         float* __restrict__ aggr, int E) {
    int e = blockIdx.x * blockDim.x + threadIdx.x;
    if (e >= E) return;
    int s = src[e], d = dst[e];
    float w = ea[e];
    atomicAdd(&aggr[d * 3 + 0], x[s * 3 + 0] * w);
    atomicAdd(&aggr[d * 3 + 1], x[s * 3 + 1] * w);
    atomicAdd(&aggr[d * 3 + 2], x[s * 3 + 2] * w);
}

template <int C>
__global__ void scatterC(const float* __restrict__ h, const int* __restrict__ src,
                         const int* __restrict__ dst, const float* __restrict__ ea,
                         float* __restrict__ aggr, int E) {
    const int CH = C / 4;                      // float4 chunks per edge
    int tid = blockIdx.x * blockDim.x + threadIdx.x;
    if (tid >= E * CH) return;
    int e  = tid / CH;
    int c4 = (tid - e * CH) * 4;
    int s = src[e], d = dst[e];
    float w = ea[e];
    const float4 v = *(const float4*)&h[(size_t)s * C + c4];
    float* ap = &aggr[(size_t)d * C + c4];
    atomicAdd(ap + 0, v.x * w);
    atomicAdd(ap + 1, v.y * w);
    atomicAdd(ap + 2, v.z * w);
    atomicAdd(ap + 3, v.w * w);
}

// ---------------------------------------------------------------------------
// layer 1 dense: K=3 is too skinny for WMMA -> scalar FMA per (node,out)
// ---------------------------------------------------------------------------
__global__ void layer1_dense(const float* __restrict__ aggr, const float* __restrict__ invc,
                             const float* __restrict__ W1, const float* __restrict__ b1,
                             float* __restrict__ h1, int N) {
    int tid = blockIdx.x * blockDim.x + threadIdx.x;
    if (tid >= N * 64) return;
    int n = tid >> 6;
    int o = tid & 63;
    float ic = invc[n];
    float v = b1[o];
#pragma unroll
    for (int k = 0; k < 3; ++k) v += aggr[n * 3 + k] * ic * W1[k * 64 + o];
    h1[tid] = fmaxf(v, 0.0f);
}

// ---------------------------------------------------------------------------
// WMMA GEMM: out = relu((A .* invc[m]) @ W + bias)
//   A:  [Mpad,K] fp32 (padded rows are zero; invc pad rows zero)
//   Wt: [N,K]    bf16 (pre-transposed) -> per-lane contiguous B loads
//   NT n-tiles per wave: A fragment loaded once, NT WMMAs issued against it.
//   POOL=true: atomically accumulate relu output into gsum[batch[m]] (fused pool)
// ---------------------------------------------------------------------------
template <int K, int N, int NT, bool POOL>
__global__ void gemm_bf16_wmma(const float* __restrict__ A, const float* __restrict__ invc,
                               const __bf16* __restrict__ Wt, const float* __restrict__ bias,
                               float* __restrict__ out, const int* __restrict__ batch,
                               int M, int Mpad) {
    const int tilesG = N / (16 * NT);         // groups of NT n-tiles
    const int tilesM = Mpad / 16;
    int wid  = (blockIdx.x * blockDim.x + threadIdx.x) >> 5;   // wave32
    int lane = threadIdx.x & 31;
    if (wid >= tilesM * tilesG) return;       // whole-wave uniform exit
    int tm = wid / tilesG;
    int tg = wid - tm * tilesG;

    int mr  = tm * 16 + (lane & 15);          // A row held by this lane (< Mpad)
    int kb  = (lane >> 4) * 8;                // per-lane K sub-base (ISA 16-bit A layout)
    int nc0 = tg * (16 * NT) + (lane & 15);   // first B col held by this lane

    float ic = invc[mr];
    const float* arow = A + (size_t)mr * K;

    v8f acc[NT];
#pragma unroll
    for (int t = 0; t < NT; ++t) acc[t] = (v8f){};

#pragma unroll
    for (int k0 = 0; k0 < K; k0 += 32) {
        // ---- A fragment: 4 unconditional float4 loads + scale + cvt ----
        const float4 f0 = *(const float4*)&arow[k0 + kb];
        const float4 f1 = *(const float4*)&arow[k0 + kb + 4];
        const float4 f2 = *(const float4*)&arow[k0 + kb + 16];
        const float4 f3 = *(const float4*)&arow[k0 + kb + 20];
        v16bf a;
        a[0]  = (__bf16)(f0.x * ic); a[1]  = (__bf16)(f0.y * ic);
        a[2]  = (__bf16)(f0.z * ic); a[3]  = (__bf16)(f0.w * ic);
        a[4]  = (__bf16)(f1.x * ic); a[5]  = (__bf16)(f1.y * ic);
        a[6]  = (__bf16)(f1.z * ic); a[7]  = (__bf16)(f1.w * ic);
        a[8]  = (__bf16)(f2.x * ic); a[9]  = (__bf16)(f2.y * ic);
        a[10] = (__bf16)(f2.z * ic); a[11] = (__bf16)(f2.w * ic);
        a[12] = (__bf16)(f3.x * ic); a[13] = (__bf16)(f3.y * ic);
        a[14] = (__bf16)(f3.z * ic); a[15] = (__bf16)(f3.w * ic);

        // ---- NT B fragments (contiguous 16B bf16 loads), NT WMMAs ----
#pragma unroll
        for (int t = 0; t < NT; ++t) {
            const __bf16* wcol = Wt + (size_t)(nc0 + 16 * t) * K + k0 + kb;
            const v8bf lo = *(const v8bf*)wcol;
            const v8bf hi = *(const v8bf*)(wcol + 16);
            v16bf b = __builtin_shufflevector(lo, hi, 0, 1, 2, 3, 4, 5, 6, 7,
                                              8, 9, 10, 11, 12, 13, 14, 15);
            acc[t] = __builtin_amdgcn_wmma_f32_16x16x32_bf16(
                /*neg_a=*/false, a, /*neg_b=*/false, b,
                /*c_mod=*/(short)0, acc[t], /*reuse_a=*/false, /*reuse_b=*/false);
        }
    }

#pragma unroll
    for (int t = 0; t < NT; ++t) {
        int nc = nc0 + 16 * t;
        float bv = bias[nc];
#pragma unroll
        for (int r = 0; r < 8; ++r) {
            int m = tm * 16 + ((lane < 16) ? r : (8 + r));   // D layout
            float v = fmaxf(acc[t][r] + bv, 0.0f);
            if (POOL) {
                if (m < M) atomicAdd(&out[(size_t)batch[m] * N + nc], v);
            } else {
                out[(size_t)m * N + nc] = v;                 // padded h buffer
            }
        }
    }
}

// ---------------------------------------------------------------------------
// final: out[g] = (sum_c gsum[g][c] * Wf[c]) / max(gcnt[g],1) + bf
// ---------------------------------------------------------------------------
__global__ void final_linear(const float* __restrict__ gsum, const float* __restrict__ gcnt,
                             const float* __restrict__ Wf, const float* __restrict__ bf_,
                             float* __restrict__ out, int G, int C) {
    int g = blockIdx.x * blockDim.x + threadIdx.x;
    if (g >= G) return;
    float s = 0.0f;
    for (int c = 0; c < C; ++c) s += gsum[g * C + c] * Wf[c];
    out[g] = s / fmaxf(gcnt[g], 1.0f) + bf_[0];
}

// ---------------------------------------------------------------------------
extern "C" void kernel_launch(void* const* d_in, const int* in_sizes, int n_in,
                              void* d_out, int out_size, void* d_ws, size_t ws_size,
                              hipStream_t stream) {
    const float* x   = (const float*)d_in[0];
    const int*   ei  = (const int*)d_in[1];
    const float* ea  = (const float*)d_in[2];
    const int*   bat = (const int*)d_in[3];
    const float* W1  = (const float*)d_in[4];
    const float* b1  = (const float*)d_in[5];
    const float* W2  = (const float*)d_in[6];
    const float* b2  = (const float*)d_in[7];
    const float* W3  = (const float*)d_in[8];
    const float* b3  = (const float*)d_in[9];
    const float* Wf  = (const float*)d_in[10];
    const float* bfv = (const float*)d_in[11];
    float* out = (float*)d_out;

    const int N    = in_sizes[0] / 3;         // 50000 nodes
    const int E    = in_sizes[2];             // 800000 edges
    const int G    = 64;                      // graphs
    const int Mpad = (N + 15) & ~15;          // padded node count (16-row tiles)

    const int* src = ei;
    const int* dst = ei + E;

    // workspace carve-out (256B-aligned regions)
    char*  base = (char*)d_ws;
    size_t off  = 0;
    auto carve = [&](size_t nbytes) -> void* {
        void* p = base + off;
        off = (off + nbytes + 255) & ~(size_t)255;
        return p;
    };
    float*  cnt  = (float*)carve((size_t)N * 4);
    float*  invc = (float*)carve((size_t)Mpad * 4);
    float*  gcnt = (float*)carve((size_t)G * 4);
    float*  gsum = (float*)carve((size_t)G * 256 * 4);
    float*  aggr = (float*)carve((size_t)Mpad * 128 * 4);   // reused: 3/64/128 ch
    float*  h1   = (float*)carve((size_t)Mpad * 64 * 4);
    float*  h2   = (float*)carve((size_t)Mpad * 128 * 4);
    __bf16* w2t  = (__bf16*)carve((size_t)64 * 128 * 2);
    __bf16* w3t  = (__bf16*)carve((size_t)128 * 256 * 2);
    (void)ws_size;

    const int TPB = 256;
    const int tilesM = Mpad / 16;

    // degrees + graph counts + bf16 weight transposes (once per launch)
    hipMemsetAsync(cnt, 0, (size_t)N * 4, stream);
    hipMemsetAsync(gcnt, 0, (size_t)G * 4, stream);
    count_edges<<<(E + TPB - 1) / TPB, TPB, 0, stream>>>(dst, cnt, E);
    inv_counts<<<(Mpad + TPB - 1) / TPB, TPB, 0, stream>>>(cnt, invc, N, Mpad);
    graph_counts<<<(N + TPB - 1) / TPB, TPB, 0, stream>>>(bat, gcnt, N);
    convert_w<<<(64 * 128 + TPB - 1) / TPB, TPB, 0, stream>>>(W2, w2t, 64, 128);
    convert_w<<<(128 * 256 + TPB - 1) / TPB, TPB, 0, stream>>>(W3, w3t, 128, 256);

    // ---- layer 1: 3 -> 64 -------------------------------------------------
    hipMemsetAsync(aggr, 0, (size_t)N * 3 * 4, stream);
    scatter3<<<(E + TPB - 1) / TPB, TPB, 0, stream>>>(x, src, dst, ea, aggr, E);
    layer1_dense<<<(N * 64 + TPB - 1) / TPB, TPB, 0, stream>>>(aggr, invc, W1, b1, h1, N);

    // ---- layer 2: 64 -> 128 (WMMA) ----------------------------------------
    hipMemsetAsync(aggr, 0, (size_t)Mpad * 64 * 4, stream);
    scatterC<64><<<(E * 16 + TPB - 1) / TPB, TPB, 0, stream>>>(h1, src, dst, ea, aggr, E);
    {
        int waves  = tilesM * (128 / (16 * 4));
        int blocks = (waves + 3) / 4;           // 128 threads = 4 waves/block
        gemm_bf16_wmma<64, 128, 4, false><<<blocks, 128, 0, stream>>>(
            aggr, invc, w2t, b2, h2, nullptr, N, Mpad);
    }

    // ---- layer 3: 128 -> 256 (WMMA, fused global mean-pool) ---------------
    hipMemsetAsync(aggr, 0, (size_t)Mpad * 128 * 4, stream);
    scatterC<128><<<(E * 32 + TPB - 1) / TPB, TPB, 0, stream>>>(h2, src, dst, ea, aggr, E);
    hipMemsetAsync(gsum, 0, (size_t)G * 256 * 4, stream);
    {
        int waves  = tilesM * (256 / (16 * 4));
        int blocks = (waves + 3) / 4;
        gemm_bf16_wmma<128, 256, 4, true><<<blocks, 128, 0, stream>>>(
            aggr, invc, w3t, b3, gsum, bat, N, Mpad);
    }

    // ---- readout ----------------------------------------------------------
    final_linear<<<1, 64, 0, stream>>>(gsum, gcnt, Wf, bfv, out, G, 256);
}
